// PConvModule_16484084483760
// MI455X (gfx1250) — matrix operand order repaired and checked by reading, not compiled, not run on hardware
//
#include <hip/hip_runtime.h>
#include <hip/hip_bf16.h>
#include <math.h>

#define CH 256
#define EPS_BN 1e-5f

typedef __attribute__((ext_vector_type(16))) __bf16 v16bf;
typedef __attribute__((ext_vector_type(8)))  float  v8f;

__device__ __forceinline__ unsigned short f2bf(float f) {
    union { float f; unsigned int u; } x; x.f = f;
    unsigned int u = x.u + 0x7FFFu + ((x.u >> 16) & 1u);  // RNE (NaN ignored)
    return (unsigned short)(u >> 16);
}

// ---------------------------------------------------------------------------
// Pre-swizzle dcn weights (Cout=256, K=2304) fp32 -> bf16 in WMMA A-fragment
// order: dst[((m*72 + kc)*32 + lane)*16 + e], contiguous k order (not the ISA table order), matching B:
//   lane<16 : K {0..15};  lane>=16 : K {16..31}
// ---------------------------------------------------------------------------
__global__ void wconv_kernel(const float* __restrict__ w,
                             unsigned short* __restrict__ dst) {
    int i = blockIdx.x * blockDim.x + threadIdx.x;   // 589824 total
    int e    = i & 15;
    int lane = (i >> 4) & 31;
    int kc   = (i >> 9) % 72;
    int m    = (i >> 9) / 72;
    int mr   = (lane < 16) ? lane : lane - 16;
    int koff = (lane < 16) ? e : e + 16;
    int K  = kc * 32 + koff;
    int co = m * 16 + mr;
    dst[i] = f2bf(w[co * 2304 + K]);
}

// Same swizzle for the offset-conv weights (18,256,3,3), Cout padded to 32.
__global__ void wconv_off_kernel(const float* __restrict__ w,
                                 unsigned short* __restrict__ dst) {
    int i = blockIdx.x * blockDim.x + threadIdx.x;   // 73728 total
    int e    = i & 15;
    int lane = (i >> 4) & 31;
    int kc   = (i >> 9) % 72;
    int m    = (i >> 9) / 72;                        // 0..1
    int mr   = (lane < 16) ? lane : lane - 16;
    int koff = (lane < 16) ? e : e + 16;
    int K  = kc * 32 + koff;
    int co = m * 16 + mr;
    dst[i] = (co < 18) ? f2bf(w[co * 2304 + K]) : (unsigned short)0;
}

// ---------------------------------------------------------------------------
// Offset conv as im2col + WMMA: 3x3 pad 1 stride s, 256 -> 18 (padded 32).
// Block: 256 threads, 32-pixel tile. Input panel staged once into LDS
// (1 gather per (cin,tap,pixel)); waves 0-1 run the 2 M-tiles of WMMA.
// ---------------------------------------------------------------------------
__global__ __launch_bounds__(256) void offgemm_kernel(
        const float* __restrict__ x, const unsigned short* __restrict__ wbfo,
        const float* __restrict__ bias, float* __restrict__ off,
        int H, int W, int Ho, int Wo, int stride, int S) {
    __shared__ __align__(32) unsigned short bmat[32 * 576];
    __shared__ int   sIdx[288];
    __shared__ float sMsk[288];

    const int tid = threadIdx.x;
    const int tilesPer = S >> 5;
    const int n  = blockIdx.x / tilesPer;
    const int p0 = (blockIdx.x % tilesPer) << 5;
    const int HW = H * W;

    for (int t = tid; t < 288; t += 256) {
        int tap = t >> 5, pixl = t & 31;
        int p  = p0 + pixl;
        int oy = p / Wo, ox = p % Wo;
        int ky = tap / 3, kx = tap % 3;
        int iy = oy * stride - 1 + ky;
        int ix = ox * stride - 1 + kx;
        bool valid = (iy >= 0) && (iy < H) && (ix >= 0) && (ix < W);
        sIdx[t] = valid ? (iy * W + ix) : 0;
        sMsk[t] = valid ? 1.f : 0.f;
    }
    __syncthreads();

    v8f c0 = {}, c1 = {};
    const int wave = tid >> 5, lane = tid & 31;
    const int colbase = (lane < 16) ? 0 : 16;
    const int ldsb = (lane & 15) * 576 + colbase;

    for (int h = 0; h < 4; ++h) {
        for (int it = 0; it < 72; ++it) {            // 72*256 == 18432
            int s   = it * 256 + tid;
            int cl  = s / 288;
            int rem = s - cl * 288;
            int tap = rem >> 5, pixl = rem & 31;
            int cin = h * 64 + cl;
            const float* xc = x + (size_t)(n * CH + cin) * HW;
            float v = sMsk[rem] * xc[sIdx[rem]];
            bmat[pixl * 576 + cl * 9 + tap] = f2bf(v);
        }
        __syncthreads();
        if (wave < 2) {                              // wave-uniform: EXEC all-1s
            for (int kcl = 0; kcl < 18; ++kcl) {
                int kc = h * 18 + kcl;
                v16bf a  = *(const v16bf*)(wbfo + ((size_t)(wave * 72 + kc) * 32 + lane) * 16);
                v16bf b0 = *(const v16bf*)(&bmat[ldsb + kcl * 32]);
                v16bf b1 = *(const v16bf*)(&bmat[ldsb + 9216 + kcl * 32]);
                c0 = __builtin_amdgcn_wmma_f32_16x16x32_bf16(false, a, false, b0, (short)0, c0, false, false);
                c1 = __builtin_amdgcn_wmma_f32_16x16x32_bf16(false, a, false, b1, (short)0, c1, false, false);
            }
        }
        __syncthreads();
    }

    if (wave < 2) {
        const int col  = lane & 15;
        const int rofs = (lane < 16) ? 0 : 8;
#pragma unroll
        for (int r = 0; r < 8; ++r) {
            int co = wave * 16 + r + rofs;
            if (co < 18) {
                float bz = bias[co];
                float* pb = off + (size_t)(n * 18 + co) * S + p0;
                pb[col]      = c0[r] + bz;
                pb[16 + col] = c1[r] + bz;
            }
        }
    }
}

// ---------------------------------------------------------------------------
// Fused deformable sampling + bf16 WMMA GEMM.
// Block: 256 threads (8 waves), 32-pixel tile x 256 output channels.
// K = 2304 processed in 4 chunks of 576 (64 input channels each).
// ---------------------------------------------------------------------------
__global__ __launch_bounds__(256) void dgemm_kernel(
        const float* __restrict__ x, const float* __restrict__ off,
        const unsigned short* __restrict__ wbf, float* __restrict__ out,
        int H, int W, int Ho, int Wo, int stride, int S, int acc_flag) {
    __shared__ __align__(32) unsigned short bmat[32 * 576];  // [pix][k_local], bf16
    __shared__ int   sIdx[288][4];                            // 9 taps * 32 px
    __shared__ float sWgt[288][4];

    const int tid = threadIdx.x;
    const int tilesPer = S >> 5;
    const int n  = blockIdx.x / tilesPer;
    const int p0 = (blockIdx.x % tilesPer) << 5;
    const int HW = H * W;

    // ---- Stage A: bilinear tap descriptors (shared across all channels) ----
    for (int t = tid; t < 288; t += 256) {
        int tap = t >> 5, pixl = t & 31;
        int p  = p0 + pixl;
        int oy = p / Wo, ox = p % Wo;
        int ky = tap / 3, kx = tap % 3;
        const float* ob = off + (size_t)((n * 18 + tap * 2) * Ho + oy) * Wo + ox;
        float dy = ob[0];
        float dx = ob[(size_t)Ho * Wo];
        float py = (float)(oy * stride - 1 + ky) + dy;
        float px = (float)(ox * stride - 1 + kx) + dx;
        float y0f = floorf(py), x0f = floorf(px);
        float wy = py - y0f, wx = px - x0f;
        int y0 = (int)y0f, x0 = (int)x0f;
        float cw[4] = { (1.f - wy) * (1.f - wx), (1.f - wy) * wx,
                        wy * (1.f - wx),         wy * wx };
        int yy[4] = { y0, y0, y0 + 1, y0 + 1 };
        int xx[4] = { x0, x0 + 1, x0, x0 + 1 };
#pragma unroll
        for (int j = 0; j < 4; ++j) {
            bool valid = (yy[j] >= 0) && (yy[j] < H) && (xx[j] >= 0) && (xx[j] < W);
            int yc = min(max(yy[j], 0), H - 1);
            int xc = min(max(xx[j], 0), W - 1);
            sIdx[t][j] = yc * W + xc;
            sWgt[t][j] = valid ? cw[j] : 0.f;
        }
    }
    __syncthreads();

    v8f c00 = {}, c01 = {}, c10 = {}, c11 = {};
    const int wave = tid >> 5, lane = tid & 31;
    const int m0 = wave, m1 = wave + 8;
    const int colbase = (lane < 16) ? 0 : 16;
    const int ldsb = (lane & 15) * 576 + colbase;   // B-fragment base (ushorts)

    for (int h = 0; h < 4; ++h) {
        // ---- Stage B: gather 64 cin * 9 taps * 32 px samples into LDS ----
        for (int it = 0; it < 72; ++it) {           // 72*256 == 18432 exactly
            int s   = it * 256 + tid;
            int cl  = s / 288;
            int rem = s - cl * 288;
            int tap = rem >> 5, pixl = rem & 31;
            int cin = h * 64 + cl;
            const float* xc = x + (size_t)(n * CH + cin) * HW;
            float v = sWgt[rem][0] * xc[sIdx[rem][0]]
                    + sWgt[rem][1] * xc[sIdx[rem][1]]
                    + sWgt[rem][2] * xc[sIdx[rem][2]]
                    + sWgt[rem][3] * xc[sIdx[rem][3]];
            bmat[pixl * 576 + cl * 9 + tap] = f2bf(v);
        }
        __syncthreads();

        // ---- Stage C: 18 k-chunks of v_wmma_f32_16x16x32_bf16 ----
        for (int kcl = 0; kcl < 18; ++kcl) {
            int kc = h * 18 + kcl;
            const unsigned short* ap0 = wbf + ((size_t)(m0 * 72 + kc) * 32 + lane) * 16;
            const unsigned short* ap1 = wbf + ((size_t)(m1 * 72 + kc) * 32 + lane) * 16;
            __builtin_prefetch(ap0 + 512, 0, 3);    // next kc's A stream
            v16bf a0 = *(const v16bf*)ap0;
            v16bf a1 = *(const v16bf*)ap1;
            v16bf b0 = *(const v16bf*)(&bmat[ldsb + kcl * 32]);
            v16bf b1 = *(const v16bf*)(&bmat[ldsb + 9216 + kcl * 32]);
            c00 = __builtin_amdgcn_wmma_f32_16x16x32_bf16(false, a0, false, b0, (short)0, c00, false, false);
            c01 = __builtin_amdgcn_wmma_f32_16x16x32_bf16(false, a0, false, b1, (short)0, c01, false, false);
            c10 = __builtin_amdgcn_wmma_f32_16x16x32_bf16(false, a1, false, b0, (short)0, c10, false, false);
            c11 = __builtin_amdgcn_wmma_f32_16x16x32_bf16(false, a1, false, b1, (short)0, c11, false, false);
        }
        __syncthreads();
    }

    // ---- Stage D: store D tiles (C/D layout: VGPR r -> M=r / M=8+r) ----
    const int col  = lane & 15;
    const int rofs = (lane < 16) ? 0 : 8;
    float* ob0 = out + (size_t)(n * CH + m0 * 16) * S + p0;
    float* ob1 = out + (size_t)(n * CH + m1 * 16) * S + p0;
#pragma unroll
    for (int r = 0; r < 8; ++r) {
        int cr = r + rofs;
        float* p00 = ob0 + (size_t)cr * S + col;
        float* p01 = p00 + 16;
        float* p10 = ob1 + (size_t)cr * S + col;
        float* p11 = p10 + 16;
        if (acc_flag) { *p00 += c00[r]; *p01 += c01[r]; *p10 += c10[r]; *p11 += c11[r]; }
        else          { *p00  = c00[r]; *p01  = c01[r]; *p10  = c10[r]; *p11  = c11[r]; }
    }
}

// ---------------------------------------------------------------------------
// Bilinear (align_corners=True) upsample of `up` (h,w) to (OH,OW), += into out.
// ---------------------------------------------------------------------------
__global__ void upsample_add_kernel(const float* __restrict__ up, float* __restrict__ out,
                                    int h, int w, int OH, int OW, int total) {
    int i = blockIdx.x * blockDim.x + threadIdx.x;
    if (i >= total) return;
    int ox = i % OW; int t = i / OW;
    int oy = t % OH; int nc = t / OH;
    float ys = oy * ((float)(h - 1) / (float)(OH - 1));
    float xs = ox * ((float)(w - 1) / (float)(OW - 1));
    int y0 = min((int)floorf(ys), h - 2);
    int x0 = min((int)floorf(xs), w - 2);
    float wy = ys - (float)y0, wx = xs - (float)x0;
    const float* u = up + (size_t)nc * h * w;
    float v00 = u[y0 * w + x0],       v01 = u[y0 * w + x0 + 1];
    float v10 = u[(y0 + 1) * w + x0], v11 = u[(y0 + 1) * w + x0 + 1];
    float r0 = v00 * (1.f - wy) + v10 * wy;
    float r1 = v01 * (1.f - wy) + v11 * wy;
    out[i] += r0 * (1.f - wx) + r1 * wx;
}

// ---------------------------------------------------------------------------
// BN (training-mode) stats over concat of 4 levels: one block per channel.
// ---------------------------------------------------------------------------
__global__ void bn_stats_kernel(const float* __restrict__ t, float* __restrict__ mean,
                                float* __restrict__ istd) {
    const int c = blockIdx.x, tid = threadIdx.x;
    __shared__ float ssum[256], ssq[256];
    const int Sl[4]   = {6144, 1536, 384, 96};
    const int base[4] = {0, 3145728, 3932160, 4128768};
    float sum = 0.f, sq = 0.f;
    for (int l = 0; l < 4; ++l)
        for (int n = 0; n < 2; ++n) {
            const float* p = t + base[l] + (size_t)(n * CH + c) * Sl[l];
            for (int s = tid; s < Sl[l]; s += 256) { float v = p[s]; sum += v; sq += v * v; }
        }
    ssum[tid] = sum; ssq[tid] = sq;
    __syncthreads();
    for (int st = 128; st > 0; st >>= 1) {
        if (tid < st) { ssum[tid] += ssum[tid + st]; ssq[tid] += ssq[tid + st]; }
        __syncthreads();
    }
    if (tid == 0) {
        float m = ssum[0] * (1.f / 16320.f);
        float v = ssq[0] * (1.f / 16320.f) - m * m;
        mean[c] = m;
        istd[c] = rsqrtf(v + EPS_BN);
    }
}

__global__ void bn_apply_kernel(float* __restrict__ out, const float* __restrict__ mean,
                                const float* __restrict__ istd, const float* __restrict__ gamma,
                                const float* __restrict__ beta, int S, int total) {
    int i = blockIdx.x * blockDim.x + threadIdx.x;
    if (i >= total) return;
    int c = (i / S) % CH;
    float v = (out[i] - mean[c]) * istd[c] * gamma[c] + beta[c];
    out[i] = v > 0.f ? v : 0.f;
}

// ---------------------------------------------------------------------------
extern "C" void kernel_launch(void* const* d_in, const int* in_sizes, int n_in,
                              void* d_out, int out_size, void* d_ws, size_t ws_size,
                              hipStream_t stream) {
    const float* x[4]     = {(const float*)d_in[0], (const float*)d_in[1],
                             (const float*)d_in[2], (const float*)d_in[3]};
    const float* off_w[3] = {(const float*)d_in[4], (const float*)d_in[7], (const float*)d_in[10]};
    const float* off_b[3] = {(const float*)d_in[5], (const float*)d_in[8], (const float*)d_in[11]};
    const float* dcn_w[3] = {(const float*)d_in[6], (const float*)d_in[9], (const float*)d_in[12]};
    const float* gamma    = (const float*)d_in[13];
    const float* beta     = (const float*)d_in[14];
    float* out = (float*)d_out;

    char* ws = (char*)d_ws;
    unsigned short* wbf[3];
    for (int i = 0; i < 3; ++i) wbf[i] = (unsigned short*)(ws + (size_t)i * 1179648);
    float* offbuf = (float*)(ws + 3538944);   //   884,736 B (max offset tensor)
    float* up1    = (float*)(ws + 4423680);   // 3,145,728 B
    float* up2    = (float*)(ws + 7569408);   //   786,432 B
    float* up3    = (float*)(ws + 8355840);   //   196,608 B
    float* meanp  = (float*)(ws + 8552448);
    float* istdp  = meanp + 256;
    unsigned short* wbfo[3];
    for (int i = 0; i < 3; ++i) wbfo[i] = (unsigned short*)(ws + 8554496 + (size_t)i * 147456);

    // 1) weights -> bf16, WMMA-swizzled (dcn full, offset padded 18->32)
    for (int i = 0; i < 3; ++i) {
        wconv_kernel<<<2304, 256, 0, stream>>>(dcn_w[i], wbf[i]);
        wconv_off_kernel<<<288, 256, 0, stream>>>(off_w[i], wbfo[i]);
    }

    const int Hs[4] = {64, 32, 16, 8}, Wsz[4] = {96, 48, 24, 12};
    const int Ss[4] = {6144, 1536, 384, 96};
    float* tptr[4] = {out, out + 3145728, out + 3932160, out + 4128768};

    struct Call { int src; int wi; int stride; float* dst; int acc; };
    Call calls[10] = {
        {0, 1, 1, tptr[0], 0}, {1, 1, 1, tptr[1], 0},
        {2, 1, 1, tptr[2], 0}, {3, 1, 1, tptr[3], 0},   // dc1 on each level
        {0, 2, 2, tptr[1], 1}, {1, 2, 2, tptr[2], 1},
        {2, 2, 2, tptr[3], 1},                          // dc2 stride-2 adds
        {1, 0, 1, up1, 0}, {2, 0, 1, up2, 0}, {3, 0, 1, up3, 0},  // dc0 for upsample
    };
    for (int ci = 0; ci < 10; ++ci) {
        const Call& cl = calls[ci];
        int H = Hs[cl.src], W = Wsz[cl.src];
        int Ho = (H - 1) / cl.stride + 1, Wo = (W - 1) / cl.stride + 1;
        int S = Ho * Wo;
        offgemm_kernel<<<2 * (S / 32), 256, 0, stream>>>(
            x[cl.src], wbfo[cl.wi], off_b[cl.wi], offbuf, H, W, Ho, Wo, cl.stride, S);
        dgemm_kernel<<<2 * (S / 32), 256, 0, stream>>>(
            x[cl.src], offbuf, wbf[cl.wi], cl.dst, H, W, Ho, Wo, cl.stride, S, cl.acc);
    }

    // 2) bilinear upsample-adds (levels 0..2 get up from level l+1)
    float* ups[3] = {up1, up2, up3};
    for (int l = 0; l < 3; ++l) {
        int tot = 2 * CH * Ss[l];
        upsample_add_kernel<<<(tot + 255) / 256, 256, 0, stream>>>(
            ups[l], tptr[l], Hs[l + 1], Wsz[l + 1], Hs[l], Wsz[l], tot);
    }

    // 3) IBN + ReLU (in place on d_out)
    bn_stats_kernel<<<256, 256, 0, stream>>>(out, meanp, istdp);
    for (int l = 0; l < 4; ++l) {
        int tot = 2 * CH * Ss[l];
        bn_apply_kernel<<<(tot + 255) / 256, 256, 0, stream>>>(
            tptr[l], meanp, istdp, gamma, beta, Ss[l], tot);
    }
}
